// QuantLinear_10548439679562
// MI455X (gfx1250) — matrix operand order repaired and checked by reading, not compiled, not run on hardware
//
#include <hip/hip_runtime.h>

typedef __attribute__((ext_vector_type(16))) __bf16        v16bf;
typedef __attribute__((ext_vector_type(8)))  float         v8f;
typedef __attribute__((ext_vector_type(4)))  unsigned int  v4u;
typedef __attribute__((ext_vector_type(2)))  unsigned int  v2u;

#define IN_F   4096
#define OUT_F  11008
#define GROUP  128
#define BM     128
#define BN     128
#define BK     64
#define KP     72                 // padded K extent in LDS (elements)
#define NSTAGE (IN_F / BK)        // 64

union FragU { v16bf v; v4u q[2]; };

// pack the two bf16 (= high halves of rounded f32 bits) into one dword:
// result = {hi[31:16], lo[31:16]}  ->  v_perm_b32
__device__ __forceinline__ unsigned int pack_bf16_hi(unsigned int hi, unsigned int lo) {
  return __builtin_amdgcn_perm(hi, lo, 0x07060302u);
}

__global__ __launch_bounds__(256)
void gptq_w4_gemm_bf16_wmma(const float* __restrict__ x,
                            const float* __restrict__ scales,
                            const float* __restrict__ bias,
                            const int*   __restrict__ qweight,
                            const int*   __restrict__ qzeros,
                            float*       __restrict__ out) {
  __shared__ unsigned short ldsX[2][BM * KP];   // x tile, bf16, [M][K]
  __shared__ unsigned short ldsW[2][BN * KP];   // weight tile, bf16, [N][K] (B^T)

  const int t    = threadIdx.x;
  const int lane = t & 31;
  const int wave = t >> 5;
  const int nrow = lane & 15;     // row-within-16 of a fragment
  const int lh   = lane >> 4;     // lane half (0/1)

  const int bm = blockIdx.y * BM;
  const int bn = blockIdx.x * BN;

  // 8 waves in a 4(M) x 2(N) arrangement; wave tile = 32(M) x 64(N)
  const int wm = (wave & 3) * 32;
  const int wn = (wave >> 2) * 64;

  // ---- x global load mapping: 128 rows x 16 float4 per stage ----
  const int xc4 = t & 15;         // which float4 (of 16) along K
  const int xr0 = t >> 4;         // base row; rows xr0 + 16*i, i=0..7

  // ---- weight load mapping: one N column per thread, 4 packed-K rows ----
  const int wn_loc = t & 127;     // local N (0..127)
  const int wk0    = t >> 7;      // 0/1 ; packed-k rows wk0 + 2*i, i=0..3

  // prefetch registers
  float4       xr[8];
  unsigned int qr[4];
  unsigned int spk;   // bf16(scale) duplicated into both halves
  unsigned int mzpk;  // bf16(-(128+zero)) duplicated into both halves

  v8f c[2][4];
  #pragma unroll
  for (int i = 0; i < 2; ++i)
    #pragma unroll
    for (int j = 0; j < 4; ++j)
      #pragma unroll
      for (int r = 0; r < 8; ++r)
        c[i][j][r] = 0.0f;

  auto loadGlobal = [&](int s) {
    const int k0 = s * BK;
    const float* xp = x + (size_t)(bm + xr0) * IN_F + k0 + xc4 * 4;
    #pragma unroll
    for (int i = 0; i < 8; ++i)
      xr[i] = *(const float4*)(xp + (size_t)(16 * i) * IN_F);

    const int n   = bn + wn_loc;
    const int kpB = s * (BK / 8);
    #pragma unroll
    for (int i = 0; i < 4; ++i)
      qr[i] = (unsigned int)qweight[(size_t)(kpB + wk0 + 2 * i) * OUT_F + n];

    const int g = k0 / GROUP;
    // packed bf16 scale (round-to-nearest on the cast)
    unsigned int sb = (__float_as_uint(scales[(size_t)g * OUT_F + n]) + 0x8000u) >> 16;
    spk = sb | (sb << 16);
    // packed bf16 of -(128+z): 0xC300|z encodes -(128+z) exactly for z in [0,15]
    unsigned int z =
        ((unsigned int)qzeros[(size_t)g * (OUT_F / 8) + (n >> 3)] >> ((n & 7) * 4)) & 15u;
    unsigned int mz = 0xC300u | z;
    mzpk = mz | (mz << 16);
  };

  auto storeLds = [&](int buf) {
    // ---- x: fp32 -> bf16 pairs via +0x8000 round and v_perm_b32 pack ----
    #pragma unroll
    for (int i = 0; i < 8; ++i) {
      const unsigned int b0 = __float_as_uint(xr[i].x) + 0x8000u;
      const unsigned int b1 = __float_as_uint(xr[i].y) + 0x8000u;
      const unsigned int b2 = __float_as_uint(xr[i].z) + 0x8000u;
      const unsigned int b3 = __float_as_uint(xr[i].w) + 0x8000u;
      v2u d;
      d[0] = pack_bf16_hi(b1, b0);
      d[1] = pack_bf16_hi(b3, b2);
      *(v2u*)&ldsX[buf][(xr0 + 16 * i) * KP + xc4 * 4] = d;
    }
    // ---- weights: magic-bias int4 -> bf16, packed bf16 affine dequant ----
    // qpk = {128+q1, 128+q0} (exact bf16); (qpk + mzpk) is exact (integers
    // in [-15,15]); final v_pk_mul_bf16 is the single rounding step.
    #pragma unroll
    for (int i = 0; i < 4; ++i) {
      const unsigned int q = qr[i];
      v4u d;
      #pragma unroll
      for (int j = 0; j < 4; ++j) {
        const unsigned int tq  = q >> (8 * j);
        const unsigned int qpk = (tq & 0x0000000Fu) | ((tq & 0x000000F0u) << 12)
                               | 0x43004300u;
        unsigned int dq, w;
        asm("v_pk_add_bf16 %0, %1, %2" : "=v"(dq) : "v"(qpk), "v"(mzpk));
        asm("v_pk_mul_bf16 %0, %1, %2" : "=v"(w)  : "v"(dq),  "v"(spk));
        d[j] = w;
      }
      *(v4u*)&ldsW[buf][wn_loc * KP + (wk0 + 2 * i) * 8] = d;
    }
  };

  auto compute = [&](int buf) {
    #pragma unroll
    for (int kk = 0; kk < 2; ++kk) {          // two K=32 steps per stage
      const int k0 = kk * 32;
      FragU a[2], b[4];
      // A frag (16x32 bf16): lane half 0 -> K {0..7,16..23}, half 1 -> {8..15,24..31}
      #pragma unroll
      for (int mi = 0; mi < 2; ++mi) {
        const unsigned short* p =
            &ldsX[buf][(wm + mi * 16 + nrow) * KP + k0 + lh * 8];
        a[mi].q[0] = *(const v4u*)p;
        a[mi].q[1] = *(const v4u*)(p + 16);
      }
      // B frag (32x16 bf16, stored as B^T rows): lane half 0 -> K {0..7,16..23}, half 1 -> {8..15,24..31}
      #pragma unroll
      for (int ni = 0; ni < 4; ++ni) {
        const unsigned short* p =
            &ldsW[buf][(wn + ni * 16 + nrow) * KP + k0 + lh * 8];
        b[ni].q[0] = *(const v4u*)p;
        b[ni].q[1] = *(const v4u*)(p + 16);
      }
      #pragma unroll
      for (int mi = 0; mi < 2; ++mi)
        #pragma unroll
        for (int ni = 0; ni < 4; ++ni)
          c[mi][ni] = __builtin_amdgcn_wmma_f32_16x16x32_bf16(
              false, a[mi].v, false, b[ni].v, (short)0, c[mi][ni], false, false);
    }
  };

  // ---- pipelined main loop: global prefetch s+1 overlapped with compute s ----
  loadGlobal(0);
  storeLds(0);
  __syncthreads();
  for (int s = 0; s < NSTAGE; ++s) {
    if (s + 1 < NSTAGE) loadGlobal(s + 1);
    compute(s & 1);
    if (s + 1 < NSTAGE) storeLds((s + 1) & 1);
    __syncthreads();
  }

  // ---- epilogue: C VGPR r -> M = r (lanes 0-15) / 8+r (lanes 16-31), N = lane%16
  #pragma unroll
  for (int ni = 0; ni < 4; ++ni) {
    const int n  = bn + wn + ni * 16 + nrow;
    const float bv = bias[n];
    #pragma unroll
    for (int mi = 0; mi < 2; ++mi) {
      const int mbase = bm + wm + mi * 16 + lh * 8;
      #pragma unroll
      for (int r = 0; r < 8; ++r)
        out[(size_t)(mbase + r) * OUT_F + n] = c[mi][ni][r] + bv;
    }
  }
}

extern "C" void kernel_launch(void* const* d_in, const int* in_sizes, int n_in,
                              void* d_out, int out_size, void* d_ws, size_t ws_size,
                              hipStream_t stream) {
  const float* x       = (const float*)d_in[0];
  const float* scales  = (const float*)d_in[1];
  const float* bias    = (const float*)d_in[2];
  const int*   qweight = (const int*)  d_in[3];
  const int*   qzeros  = (const int*)  d_in[4];
  float*       out     = (float*)d_out;

  const int M = in_sizes[0] / IN_F;          // 4*2048 = 8192
  dim3 grid(OUT_F / BN, M / BM);             // (86, 64)
  gptq_w4_gemm_bf16_wmma<<<grid, 256, 0, stream>>>(x, scales, bias, qweight,
                                                   qzeros, out);
}
